// LSTM_48198122996164
// MI455X (gfx1250) — matrix operand order repaired and checked by reading, not compiled, not run on hardware
//
#include <hip/hip_runtime.h>
#include <hip/hip_bf16.h>

// Problem constants
constexpr int kE = 256;   // embedding
constexpr int kH = 128;   // hidden
constexpr int kB = 64;    // batch
constexpr int kT = 2048;  // time
constexpr int kG = 4 * kH; // 512 gate columns

typedef __attribute__((ext_vector_type(16))) _Float16 v16h;
typedef __attribute__((ext_vector_type(8)))  _Float16 h8;
typedef __attribute__((ext_vector_type(8)))  float    v8f;
typedef __attribute__((ext_vector_type(4)))  float    f4;

__device__ __forceinline__ v8f wmma16(v16h a, v16h b, v8f c) {
    // D = A(16x32 f16) * B(32x16 f16) + C(16x16 f32)
    return __builtin_amdgcn_wmma_f32_16x16x32_f16(
        false, a, false, b, (short)0, c, false, false);
}

// A fragment (16x32 f16) from a row-major f16 row pointer (LDS or global).
// ISA layout: lane L -> M = L%16; K half = (L>=16)?8:0;
// elems 0..7 -> K = k*32 + half + 0..7 ; elems 8..15 -> K = k*32 + 16 + half + 0..7
// => two aligned 16-byte loads per fragment, no conversion.
__device__ __forceinline__ v16h load_a_f16(const _Float16* rowptr, int k, int lane) {
    const int khalf = (lane >> 4) << 3;          // 0 or 8
    const _Float16* p = rowptr + k * 32 + khalf;
    h8 lo = *(const h8*)(p);
    h8 hi = *(const h8*)(p + 16);
    v16h a;
#pragma unroll
    for (int i = 0; i < 8; ++i) { a[i] = lo[i]; a[i + 8] = hi[i]; }
    return a;
}

// B fragment (32x16 f16) from a global fp32 weight row (K contiguous).
// ISA layout: lane L -> N = L%16; K half = (L>=16)?8:0; elems 0..7 -> K = half + 0..7; elems 8..15 -> K = 16 + half + 0..7
__device__ __forceinline__ v16h load_b_f32(const float* wrow, int kbase, int lane) {
    const int khalf = (lane >> 4) << 3;          // 0 or 8
    const float* p = wrow + kbase + khalf;
    v16h b;
#pragma unroll
    for (int i = 0; i < 16; ++i) b[i] = (_Float16)p[i < 8 ? i : i + 8];
    return b;
}

__device__ __forceinline__ float fsigmoid(float x) {
    return 1.f / (1.f + __expf(-x));
}
__device__ __forceinline__ float ftanh(float x) {
    return 2.f / (1.f + __expf(-2.f * x)) - 1.f;
}

// ---------------------------------------------------------------------------
// Kernel 0: one-shot f32 -> f16 conversion of text (done once, not 32x per
// consumer wave). Bandwidth bound: 134 MB read + 67 MB write ~ 9 us @ 23.3 TB/s.
// ---------------------------------------------------------------------------
__global__ __launch_bounds__(256) void cvt_kernel(
    const float* __restrict__ in, _Float16* __restrict__ outp)
{
    const size_t i = ((size_t)blockIdx.x * 256 + threadIdx.x) * 8;
    f4 a = *(const f4*)(in + i);
    f4 b = *(const f4*)(in + i + 4);
    h8 o;
#pragma unroll
    for (int j = 0; j < 4; ++j) { o[j] = (_Float16)a[j]; o[4 + j] = (_Float16)b[j]; }
    *(h8*)(outp + i) = o;
}

// ---------------------------------------------------------------------------
// Kernel A: xg[dir][t][b][g] = sum_e x16[b][t][e]*w_ih[dir][g][e] + bias[g]
// (f16 out; bias = b_ih + b_hh folded in here so the 2048-step scan never
// touches it). Wave-level GEMM: each wave owns (dir, 2 N-tiles, 256 M-tiles).
// 1024 waves = 128 blocks x 8 waves. B fragments live in registers.
// ---------------------------------------------------------------------------
__global__ __launch_bounds__(256) void proj_kernel(
    const _Float16* __restrict__ x16,
    const float* __restrict__ w_ih_f, const float* __restrict__ w_ih_b,
    const float* __restrict__ b_ih_f, const float* __restrict__ b_hh_f,
    const float* __restrict__ b_ih_b, const float* __restrict__ b_hh_b,
    _Float16* __restrict__ xg_f, _Float16* __restrict__ xg_b)
{
    const int lane  = threadIdx.x & 31;
    const int gw    = blockIdx.x * 8 + (threadIdx.x >> 5);
    const int mg    = gw & 31;          // 32 M-groups
    const int npair = (gw >> 5) & 15;   // 16 N-pairs (2 tiles each) per dir
    const int dir   = gw >> 9;          // 2 directions
    const float*    w  = dir ? w_ih_b : w_ih_f;
    _Float16*       xg = dir ? xg_b  : xg_f;
    const int n0 = npair * 2;

    // Preload 16 B fragments (2 N-tiles x 8 K-tiles) + per-lane bias
    v16h Bf[2][8];
    float bv[2];
#pragma unroll
    for (int nt = 0; nt < 2; ++nt) {
        const int gcol = (n0 + nt) * 16 + (lane & 15);
        bv[nt] = dir ? (b_ih_b[gcol] + b_hh_b[gcol]) : (b_ih_f[gcol] + b_hh_f[gcol]);
        const float* wrow = w + (size_t)gcol * kE;
#pragma unroll
        for (int k = 0; k < 8; ++k) Bf[nt][k] = load_b_f32(wrow, k * 32, lane);
    }

    const int mo = (lane >= 16) ? 8 : 0;
    for (int mi = 0; mi < 256; ++mi) {
        const int m  = mg * 256 + mi;   // M-tile id, output rows = t*64 + b
        const int t  = m >> 2;
        const int b0 = (m & 3) * 16;
        const _Float16* arow = x16 + ((size_t)(b0 + (lane & 15)) * kT + t) * kE;

        v8f C0 = {}; v8f C1 = {};
#pragma unroll
        for (int k = 0; k < 8; ++k) {
            v16h A = load_a_f16(arow, k, lane);
            C0 = wmma16(A, Bf[0][k], C0);
            C1 = wmma16(A, Bf[1][k], C1);
        }
        // D layout: lane holds N = lane%16, rows M = v + mo
        _Float16* orow = xg + (size_t)t * kB * kG;
#pragma unroll
        for (int nt = 0; nt < 2; ++nt) {
            const int gcol = (n0 + nt) * 16 + (lane & 15);
            v8f C = nt ? C1 : C0;
#pragma unroll
            for (int v = 0; v < 8; ++v)
                orow[(size_t)(b0 + v + mo) * kG + gcol] = (_Float16)(C[v] + bv[nt]);
        }
    }
}

// ---------------------------------------------------------------------------
// Kernel B: persistent sequential LSTM scan. grid = 2 (dir), block = 512 (16 waves).
// h: LDS f16 [64][136 padded]; gates: LDS f32 [64][516 padded] (no cvt round
// trip); c,h: fp32 regs. w_hh stays in registers as WMMA B fragments for all
// 2048 steps. Two barriers per step.
// ---------------------------------------------------------------------------
constexpr int HROW = 136;                       // padded f16 stride for h
constexpr int GROW = 516;                       // padded f32 stride for gates
constexpr size_t SMEM_H  = (size_t)kB * HROW * sizeof(_Float16);  //  17408
constexpr size_t SMEM_GT = (size_t)kB * GROW * sizeof(float);     // 132096
constexpr size_t SMEM_BYTES = SMEM_H + SMEM_GT;                   // 149504

__global__ __launch_bounds__(512) void lstm_rec_kernel(
    const _Float16* __restrict__ xg_f, const _Float16* __restrict__ xg_b,
    const float* __restrict__ w_hh_f, const float* __restrict__ w_hh_b,
    float* __restrict__ h_fin)
{
    extern __shared__ char smem[];
    _Float16* h_sh     = (_Float16*)smem;
    float*    gates_sh = (float*)(smem + SMEM_H);

    const int dir  = blockIdx.x;
    const _Float16* xg  = dir ? xg_b  : xg_f;
    const float*   w_hh = dir ? w_hh_b : w_hh_f;

    const int tid  = threadIdx.x;
    const int lane = tid & 31;
    const int wv   = tid >> 5;          // 16 waves

    for (int i = tid; i < kB * HROW; i += 512) h_sh[i] = (_Float16)0.f;

    // B fragments of w_hh: wave wv owns gate cols [wv*32, wv*32+32)
    const int n0 = wv * 2;
    v16h Bf[2][4];
#pragma unroll
    for (int nt = 0; nt < 2; ++nt) {
        const int gcol = (n0 + nt) * 16 + (lane & 15);
        const float* wrow = w_hh + (size_t)gcol * kH;
#pragma unroll
        for (int k = 0; k < 4; ++k) Bf[nt][k] = load_b_f32(wrow, k * 32, lane);
    }

    // Elementwise ownership: thread -> batch row eb, hidden slice [jb, jb+16)
    const int eb = tid >> 3;
    const int jb = (tid & 7) * 16;
    float creg[16], hreg[16];
#pragma unroll
    for (int i = 0; i < 16; ++i) { creg[i] = 0.f; hreg[i] = 0.f; }

    const int mo = (lane >= 16) ? 8 : 0;

    for (int s = 0; s < kT; ++s) {
        const int t = dir ? (kT - 1 - s) : s;
        __syncthreads();   // h_sh(new) ready; gates_sh free to overwrite

        // Prefetch next step's xg rows while the WMMAs run
        if (s + 1 < kT) {
            const int tn = dir ? (kT - 2 - s) : (s + 1);
            const _Float16* pf = xg + ((size_t)tn * kB + eb) * kG + jb;
            __builtin_prefetch((const void*)(pf),       0, 0);
            __builtin_prefetch((const void*)(pf + 128), 0, 0);
            __builtin_prefetch((const void*)(pf + 256), 0, 0);
            __builtin_prefetch((const void*)(pf + 384), 0, 0);
        }

        // --- GEMM phase: gates_hh = h @ w_hh.T for this wave's 32 columns ---
#pragma unroll
        for (int m = 0; m < 4; ++m) {           // 4 M-tiles cover B=64
            v8f C0 = {}; v8f C1 = {};
#pragma unroll
            for (int k = 0; k < 4; ++k) {       // K = 128
                v16h A = load_a_f16(h_sh + (m * 16 + (lane & 15)) * HROW, k, lane);
                C0 = wmma16(A, Bf[0][k], C0);
                C1 = wmma16(A, Bf[1][k], C1);
            }
            const int rbase = m * 16 + mo;
#pragma unroll
            for (int nt = 0; nt < 2; ++nt) {
                const int gcol = (n0 + nt) * 16 + (lane & 15);
                v8f C = nt ? C1 : C0;
#pragma unroll
                for (int v = 0; v < 8; ++v)
                    gates_sh[(rbase + v) * GROW + gcol] = C[v];   // f32, no cvt
            }
        }
        __syncthreads();   // gates complete

        // --- elementwise phase: gate nonlinearities + state update ---
        const _Float16* xr = xg + ((size_t)t * kB + eb) * kG + jb;  // bias pre-folded
        const float*    gr = gates_sh + eb * GROW + jb;
#pragma unroll
        for (int i = 0; i < 16; ++i) {
            const float ig = fsigmoid((float)xr[i]       + gr[i]);
            const float fg = fsigmoid((float)xr[128 + i] + gr[128 + i]);
            const float gg = ftanh   ((float)xr[256 + i] + gr[256 + i]);
            const float og = fsigmoid((float)xr[384 + i] + gr[384 + i]);
            creg[i] = fg * creg[i] + ig * gg;
            hreg[i] = og * ftanh(creg[i]);
            h_sh[eb * HROW + jb + i] = (_Float16)hreg[i];
        }
    }

    // final hidden state -> workspace
#pragma unroll
    for (int i = 0; i < 16; ++i)
        h_fin[((size_t)dir * kB + eb) * kH + jb + i] = hreg[i];
}

// ---------------------------------------------------------------------------
// Kernel C: out[b] = [h_f ; h_b] . w_out + b_out
// ---------------------------------------------------------------------------
__global__ __launch_bounds__(64) void out_kernel(
    const float* __restrict__ h_fin, const float* __restrict__ w_out,
    const float* __restrict__ b_out, float* __restrict__ out)
{
    const int b = threadIdx.x;
    float s = b_out[0];
#pragma unroll 4
    for (int j = 0; j < kH; ++j) s += h_fin[(size_t)b * kH + j] * w_out[j];
#pragma unroll 4
    for (int j = 0; j < kH; ++j) s += h_fin[(size_t)(kB + b) * kH + j] * w_out[kH + j];
    out[b] = s;
}

extern "C" void kernel_launch(void* const* d_in, const int* in_sizes, int n_in,
                              void* d_out, int out_size, void* d_ws, size_t ws_size,
                              hipStream_t stream) {
    const float* text   = (const float*)d_in[0];
    const float* w_ih_f = (const float*)d_in[1];
    const float* w_hh_f = (const float*)d_in[2];
    const float* b_ih_f = (const float*)d_in[3];
    const float* b_hh_f = (const float*)d_in[4];
    const float* w_ih_b = (const float*)d_in[5];
    const float* w_hh_b = (const float*)d_in[6];
    const float* b_ih_b = (const float*)d_in[7];
    const float* b_hh_b = (const float*)d_in[8];
    const float* w_out  = (const float*)d_in[9];
    const float* b_out  = (const float*)d_in[10];

    char* ws = (char*)d_ws;
    const size_t xg_bytes  = (size_t)kT * kB * kG * sizeof(_Float16); // 128 MiB each
    const size_t x16_elems = (size_t)kB * kT * kE;                    // 33.5 M
    _Float16* xg_f  = (_Float16*)ws;
    _Float16* xg_b  = (_Float16*)(ws + xg_bytes);
    float*    h_fin = (float*)(ws + 2 * xg_bytes);
    _Float16* x16   = (_Float16*)(ws + 2 * xg_bytes + 4096);

    // 0) text f32 -> f16, once
    cvt_kernel<<<(int)(x16_elems / (256 * 8)), 256, 0, stream>>>(text, x16);

    // 1) Input projection (+bias fold): 1024 waves, WMMA f16
    proj_kernel<<<128, 256, 0, stream>>>(x16, w_ih_f, w_ih_b,
                                         b_ih_f, b_hh_f, b_ih_b, b_hh_b,
                                         xg_f, xg_b);

    // 2) Persistent sequential scan: one workgroup per direction
    (void)hipFuncSetAttribute(reinterpret_cast<const void*>(lstm_rec_kernel),
                              hipFuncAttributeMaxDynamicSharedMemorySize,
                              (int)SMEM_BYTES);
    lstm_rec_kernel<<<2, 512, SMEM_BYTES, stream>>>(
        xg_f, xg_b, w_hh_f, w_hh_b, h_fin);

    // 3) Output head
    out_kernel<<<1, 64, 0, stream>>>(h_fin, w_out, b_out, (float*)d_out);
}
